// GAT_LSTM_Model_52604759441754
// MI455X (gfx1250) — compile-verified, matrix operand order checked by reading, not run on hardware
//
#include <hip/hip_runtime.h>
#include <hip/hip_bf16.h>

#define BB   32
#define SS   12
#define NN   200
#define CIN  2
#define HH   4
#define GHIDc 16
#define GHc  64
#define LHD  512
#define OUTD 200
#define EE   3200
#define E2c  (EE + NN)     // 3400
#define RR   (BB * SS)     // 384

typedef __attribute__((ext_vector_type(16))) __bf16 v16bf;
typedef __attribute__((ext_vector_type(8)))  __bf16 v8bf;
typedef __attribute__((ext_vector_type(8)))  float  v8f;

__device__ __forceinline__ float sigm_(float x) { return 1.0f / (1.0f + __expf(-x)); }

// ---------------------------------------------------------------------------
// Weight pack: fp32 [K,Nreal] (or [Nreal,K] if transposed) -> bf16 WMMA
// B-fragment layout: dst[((kt*NT + nt)*32 + lane)*16 + i] holds
// B[k = kt*32 + klocal(lane,i), col = nt*16 + (lane&15)]
// klocal = (i&7) + (i>>3)*16 + (lane>>4)*8     (16-bit B 32x16 layout)
// ---------------------------------------------------------------------------
__global__ void pack_b_kernel(const float* __restrict__ W, __bf16* __restrict__ dst,
                              int K, int Npad, int Nreal, int transposed) {
  int idx = blockIdx.x * 256 + threadIdx.x;
  if (idx >= K * Npad) return;
  int i    = idx & 15;
  int lane = (idx >> 4) & 31;
  int tile = idx >> 9;
  int NT   = Npad >> 4;
  int nt   = tile % NT;
  int kt   = tile / NT;
  int col  = nt * 16 + (lane & 15);
  int kl   = (i & 7) + ((i >> 3) * 16) + ((lane >> 4) * 8);
  int k    = kt * 32 + kl;
  float v = 0.0f;
  if (col < Nreal)
    v = transposed ? W[(size_t)col * K + k] : W[(size_t)k * Nreal + col];
  dst[idx] = (__bf16)v;
}

// ---------------------------------------------------------------------------
// Generic bf16 WMMA GEMM: C[M,Npad] = A[M,K] * Bpk(K,Npad)
// One wave computes a 32x64 tile: 2 A frags x 4 B frags = 8 WMMAs / k-step,
// doubling arithmetic intensity per B-fragment load (GEMM is L2-bound).
// mode 0: outH = bf16(relu(acc + v1[col]))
// mode 1: outF = acc + v1[col] + v2[col]
// mode 2: outF = acc + v1[row*addStride + col]
// mode 3: outF[row*Nguard+col] = acc + v1[col]   (col < Nguard)
// ---------------------------------------------------------------------------
__global__ __launch_bounds__(256) void wmma_gemm_kernel(
    const __bf16* __restrict__ A, const __bf16* __restrict__ Bpk,
    int M, int Npad, int K, int mode,
    const float* __restrict__ v1, const float* __restrict__ v2, int addStride,
    float* __restrict__ outF, __bf16* __restrict__ outH, int Nguard) {
  int wid  = (int)((blockIdx.x * blockDim.x + threadIdx.x) >> 5);
  int lane = threadIdx.x & 31;
  int MT = M >> 5, NGp = Npad >> 6;      // 32-row strips x 64-col groups
  if (wid >= MT * NGp) return;           // wave-uniform: EXEC stays all-ones
  int mt = wid % MT, ng = wid / MT;
  int NT = Npad >> 4;

  const __bf16* Arow0 = A + (size_t)(mt * 32 + (lane & 15)) * K + ((lane >> 4) * 8);
  const __bf16* Arow1 = Arow0 + (size_t)16 * K;
  const __bf16* Bbase = Bpk + ((size_t)(ng * 4) * 32 + lane) * 16;

  v8f acc[8];
#pragma unroll
  for (int j = 0; j < 8; ++j) acc[j] = (v8f){};

  for (int k0 = 0; k0 < K; k0 += 32) {
    v8bf a0lo = *(const v8bf*)(Arow0 + k0);
    v8bf a0hi = *(const v8bf*)(Arow0 + k0 + 16);
    v8bf a1lo = *(const v8bf*)(Arow1 + k0);
    v8bf a1hi = *(const v8bf*)(Arow1 + k0 + 16);
    v16bf a0, a1;
#pragma unroll
    for (int i = 0; i < 8; ++i) {
      a0[i] = a0lo[i]; a0[i + 8] = a0hi[i];
      a1[i] = a1lo[i]; a1[i + 8] = a1hi[i];
    }
    const __bf16* bp = Bbase + (size_t)(k0 >> 5) * NT * 512;
    v16bf b0 = *(const v16bf*)(bp);
    v16bf b1 = *(const v16bf*)(bp + 512);
    v16bf b2 = *(const v16bf*)(bp + 1024);
    v16bf b3 = *(const v16bf*)(bp + 1536);
    if (k0 + 32 < K) __builtin_prefetch(bp + (size_t)NT * 512, 0, 1);
    acc[0] = __builtin_amdgcn_wmma_f32_16x16x32_bf16(false, a0, false, b0, (short)0, acc[0], false, false);
    acc[1] = __builtin_amdgcn_wmma_f32_16x16x32_bf16(false, a0, false, b1, (short)0, acc[1], false, false);
    acc[2] = __builtin_amdgcn_wmma_f32_16x16x32_bf16(false, a0, false, b2, (short)0, acc[2], false, false);
    acc[3] = __builtin_amdgcn_wmma_f32_16x16x32_bf16(false, a0, false, b3, (short)0, acc[3], false, false);
    acc[4] = __builtin_amdgcn_wmma_f32_16x16x32_bf16(false, a1, false, b0, (short)0, acc[4], false, false);
    acc[5] = __builtin_amdgcn_wmma_f32_16x16x32_bf16(false, a1, false, b1, (short)0, acc[5], false, false);
    acc[6] = __builtin_amdgcn_wmma_f32_16x16x32_bf16(false, a1, false, b2, (short)0, acc[6], false, false);
    acc[7] = __builtin_amdgcn_wmma_f32_16x16x32_bf16(false, a1, false, b3, (short)0, acc[7], false, false);
  }

  int colb = ng * 64 + (lane & 15);
#pragma unroll
  for (int r = 0; r < 2; ++r) {
    int rowb = mt * 32 + r * 16 + ((lane >> 4) << 3);
#pragma unroll
    for (int j = 0; j < 4; ++j) {
      int col = colb + j * 16;
      v8f cc = acc[r * 4 + j];
#pragma unroll
      for (int e = 0; e < 8; ++e) {
        int row = rowb + e;
        float av = cc[e];
        if (mode == 0) {
          float vv = av + v1[col];
          outH[(size_t)row * Npad + col] = (__bf16)(vv > 0.f ? vv : 0.f);
        } else if (mode == 1) {
          outF[(size_t)row * Npad + col] = av + v1[col] + v2[col];
        } else if (mode == 2) {
          outF[(size_t)row * Npad + col] = av + v1[(size_t)row * addStride + col];
        } else {
          if (col < Nguard) outF[(size_t)row * Nguard + col] = av + v1[col];
        }
      }
    }
  }
}

// ---------------------------------------------------------------------------
// GAT: edge preprocessing (reroute self loops, append loops), CSR by dst
// ---------------------------------------------------------------------------
__global__ void edge_build_kernel(const int* __restrict__ ei, int* src_a, int* dst_a) {
  int e = blockIdx.x * 256 + threadIdx.x;
  if (e >= E2c) return;
  if (e < EE) {
    int s = ei[e], d = ei[EE + e];
    if (s == d) s = (s + 1) % NN;
    src_a[e] = s; dst_a[e] = d;
  } else { src_a[e] = e - EE; dst_a[e] = e - EE; }
}
__global__ void zero_cnt_kernel(int* cnt) {
  int i = threadIdx.x; if (i < NN) cnt[i] = 0;
}
__global__ void count_kernel(const int* __restrict__ dst_a, int* cnt) {
  int e = blockIdx.x * 256 + threadIdx.x;
  if (e < E2c) atomicAdd(&cnt[dst_a[e]], 1);
}
__global__ void scan_kernel(const int* __restrict__ cnt, int* ofs, int* fill) {
  int acc = 0;
  for (int i = 0; i < NN; ++i) { ofs[i] = acc; fill[i] = acc; acc += cnt[i]; }
  ofs[NN] = acc;
}
__global__ void scatter_kernel(const int* __restrict__ dst_a, int* fill, int* eix) {
  int e = blockIdx.x * 256 + threadIdx.x;
  if (e < E2c) { int p = atomicAdd(&fill[dst_a[e]], 1); eix[p] = e; }
}

// h = x @ W_gat[s]; a_s/a_d per head (CIN=2 so scalar FMA)
__global__ void gat_h_kernel(const float* __restrict__ x, const float* __restrict__ Wg,
                             const float* __restrict__ asr, const float* __restrict__ adt,
                             float* __restrict__ hmat, float* __restrict__ a_s,
                             float* __restrict__ a_d) {
  int tid = blockIdx.x * 256 + threadIdx.x;
  if (tid >= RR * NN) return;
  int bs = tid / NN;
  int s = bs % SS;
  const float* xp = x + (size_t)tid * CIN;
  float x0 = xp[0], x1 = xp[1];
  const float* W0 = Wg + (size_t)s * CIN * GHc;
  const float* av = asr + s * GHc;
  const float* dv = adt + s * GHc;
  float* hp = hmat + (size_t)tid * GHc;
  float accs[HH] = {0, 0, 0, 0}, accd[HH] = {0, 0, 0, 0};
  for (int o = 0; o < GHc; ++o) {
    float hv = x0 * W0[o] + x1 * W0[GHc + o];
    hp[o] = hv;
    accs[o >> 4] += hv * av[o];
    accd[o >> 4] += hv * dv[o];
  }
  for (int h = 0; h < HH; ++h) {
    a_s[(size_t)tid * HH + h] = accs[h];
    a_d[(size_t)tid * HH + h] = accd[h];
  }
}

// per (b,s,n,h): segment softmax over incoming edges + weighted aggregation
__global__ void gat_agg_kernel(const int* __restrict__ src_a, const int* __restrict__ ofs,
                               const int* __restrict__ eix, const float* __restrict__ a_s,
                               const float* __restrict__ a_d, const float* __restrict__ hmat,
                               const float* __restrict__ b_gat, __bf16* __restrict__ g0) {
  int tid = blockIdx.x * 256 + threadIdx.x;
  if (tid >= RR * NN * HH) return;
  int h  = tid & 3;
  int n  = (tid >> 2) % NN;
  int bs = tid / (NN * HH);
  int s  = bs % SS;
  int base = bs * NN;
  float ad = a_d[(size_t)(base + n) * HH + h];
  int e0 = ofs[n], e1 = ofs[n + 1];
  float m = -1e30f;
  for (int j = e0; j < e1; ++j) {
    int sn = src_a[eix[j]];
    float ev = a_s[(size_t)(base + sn) * HH + h] + ad;
    ev = ev > 0.f ? ev : 0.2f * ev;
    m = fmaxf(m, ev);
  }
  float den = 0.f;
  for (int j = e0; j < e1; ++j) {
    int sn = src_a[eix[j]];
    float ev = a_s[(size_t)(base + sn) * HH + h] + ad;
    ev = ev > 0.f ? ev : 0.2f * ev;
    den += __expf(ev - m);
  }
  float inv = 1.0f / (den + 1e-16f);
  float acc[GHIDc];
#pragma unroll
  for (int c = 0; c < GHIDc; ++c) acc[c] = 0.f;
  for (int j = e0; j < e1; ++j) {
    int sn = src_a[eix[j]];
    float ev = a_s[(size_t)(base + sn) * HH + h] + ad;
    ev = ev > 0.f ? ev : 0.2f * ev;
    float alpha = __expf(ev - m) * inv;
    const float* hp = hmat + (size_t)(base + sn) * GHc + h * GHIDc;
#pragma unroll
    for (int c = 0; c < GHIDc; ++c) acc[c] += alpha * hp[c];
  }
  const float* bg = b_gat + s * GHc + h * GHIDc;
  __bf16* gp = g0 + (size_t)bs * (NN * GHc) + n * GHc + h * GHIDc;
#pragma unroll
  for (int c = 0; c < GHIDc; ++c) {
    float vv = acc[c] + bg[c];
    gp[c] = (__bf16)(vv > 0.f ? vv : 0.f);
  }
}

// ---------------------------------------------------------------------------
// LSTM cell (gate order i,f,g,o); writes bf16 h for next recurrent GEMM
// ---------------------------------------------------------------------------
__global__ void lstm_cell_kernel(const float* __restrict__ gates, float* __restrict__ c,
                                 __bf16* __restrict__ hbf, __bf16* __restrict__ Hseq, int t) {
  int tid = blockIdx.x * 256 + threadIdx.x;
  if (tid >= BB * LHD) return;
  int j = tid % LHD, b = tid / LHD;
  const float* gr = gates + (size_t)b * 4 * LHD;
  float gi = gr[j], gf = gr[LHD + j], gg = gr[2 * LHD + j], go = gr[3 * LHD + j];
  float cn = sigm_(gf) * c[tid] + sigm_(gi) * tanhf(gg);
  float hn = sigm_(go) * tanhf(cn);
  c[tid] = cn;
  hbf[tid] = (__bf16)hn;
  if (Hseq) Hseq[((size_t)b * SS + t) * LHD + j] = (__bf16)hn;
}

__global__ void init_state_kernel(float* c0, float* c1, __bf16* h0, __bf16* h1) {
  int i = blockIdx.x * 256 + threadIdx.x;
  if (i < BB * LHD) { c0[i] = 0.f; c1[i] = 0.f; h0[i] = (__bf16)0.f; h1[i] = (__bf16)0.f; }
}

// ---------------------------------------------------------------------------
static void launch_gemm(hipStream_t st, const __bf16* A, const __bf16* Bpk, int M, int Npad,
                        int K, int mode, const float* v1, const float* v2, int addStride,
                        float* outF, __bf16* outH, int Nguard) {
  int tiles = (M >> 5) * (Npad >> 6);
  int blocks = (tiles + 7) / 8;
  wmma_gemm_kernel<<<blocks, 256, 0, st>>>(A, Bpk, M, Npad, K, mode, v1, v2, addStride,
                                           outF, outH, Nguard);
}

extern "C" void kernel_launch(void* const* d_in, const int* in_sizes, int n_in,
                              void* d_out, int out_size, void* d_ws, size_t ws_size,
                              hipStream_t stream) {
  (void)in_sizes; (void)n_in; (void)out_size; (void)ws_size;
  const float* x       = (const float*)d_in[0];
  const int*   ei      = (const int*)d_in[1];
  const float* W_gat   = (const float*)d_in[2];
  const float* att_src = (const float*)d_in[3];
  const float* att_dst = (const float*)d_in[4];
  const float* b_gat   = (const float*)d_in[5];
  const float* W_proj  = (const float*)d_in[6];
  const float* b_proj  = (const float*)d_in[7];
  const float* Wih0    = (const float*)d_in[8];
  const float* Whh0    = (const float*)d_in[9];
  const float* bih0    = (const float*)d_in[10];
  const float* bhh0    = (const float*)d_in[11];
  const float* Wih1    = (const float*)d_in[12];
  const float* Whh1    = (const float*)d_in[13];
  const float* bih1    = (const float*)d_in[14];
  const float* bhh1    = (const float*)d_in[15];
  const float* W_out   = (const float*)d_in[16];
  const float* b_out   = (const float*)d_in[17];
  float* out = (float*)d_out;

  char* ws = (char*)d_ws;
  size_t off = 0;
  auto alloc = [&](size_t bytes) -> char* {
    char* p = ws + off;
    off = (off + bytes + 255) & ~(size_t)255;
    return p;
  };
  int* src_a = (int*)alloc(E2c * 4);
  int* dst_a = (int*)alloc(E2c * 4);
  int* cnt   = (int*)alloc(NN * 4);
  int* ofs   = (int*)alloc((NN + 1) * 4);
  int* fill  = (int*)alloc(NN * 4);
  int* eix   = (int*)alloc(E2c * 4);
  float*  hmat = (float*)alloc((size_t)RR * NN * GHc * 4);
  float*  a_s  = (float*)alloc((size_t)RR * NN * HH * 4);
  float*  a_d  = (float*)alloc((size_t)RR * NN * HH * 4);
  __bf16* g0   = (__bf16*)alloc((size_t)RR * NN * GHc * 2);
  __bf16* Wproj_pk = (__bf16*)alloc((size_t)12800 * 3200 * 2);
  __bf16* g1       = (__bf16*)alloc((size_t)RR * 3200 * 2);
  __bf16* Wih0_pk  = (__bf16*)alloc((size_t)3200 * 2048 * 2);
  __bf16* Whh0_pk  = (__bf16*)alloc((size_t)512 * 2048 * 2);
  __bf16* Wih1_pk  = (__bf16*)alloc((size_t)512 * 2048 * 2);
  __bf16* Whh1_pk  = (__bf16*)alloc((size_t)512 * 2048 * 2);
  __bf16* Wout_pk  = (__bf16*)alloc((size_t)512 * 256 * 2);
  float*  X0    = (float*)alloc((size_t)RR * 2048 * 4);
  float*  X1    = (float*)alloc((size_t)RR * 2048 * 4);
  __bf16* H1    = (__bf16*)alloc((size_t)RR * LHD * 2);
  float*  gates = (float*)alloc((size_t)BB * 2048 * 4);
  float*  c0    = (float*)alloc(BB * LHD * 4);
  float*  c1    = (float*)alloc(BB * LHD * 4);
  __bf16* h0b   = (__bf16*)alloc(BB * LHD * 2);
  __bf16* h1b   = (__bf16*)alloc(BB * LHD * 2);

  auto cdiv = [](int a, int b) { return (a + b - 1) / b; };

  // --- Weight down-convert + WMMA-fragment pack (bf16) ---
  pack_b_kernel<<<cdiv(12800 * 3200, 256), 256, 0, stream>>>(W_proj, Wproj_pk, 12800, 3200, 3200, 0);
  pack_b_kernel<<<cdiv(3200 * 2048, 256), 256, 0, stream>>>(Wih0, Wih0_pk, 3200, 2048, 2048, 1);
  pack_b_kernel<<<cdiv(512 * 2048, 256), 256, 0, stream>>>(Whh0, Whh0_pk, 512, 2048, 2048, 1);
  pack_b_kernel<<<cdiv(512 * 2048, 256), 256, 0, stream>>>(Wih1, Wih1_pk, 512, 2048, 2048, 1);
  pack_b_kernel<<<cdiv(512 * 2048, 256), 256, 0, stream>>>(Whh1, Whh1_pk, 512, 2048, 2048, 1);
  pack_b_kernel<<<cdiv(512 * 256, 256), 256, 0, stream>>>(W_out, Wout_pk, 512, 256, 200, 0);

  // --- GAT: edges -> CSR -> h/attention -> softmax aggregate (bf16 out) ---
  edge_build_kernel<<<cdiv(E2c, 256), 256, 0, stream>>>(ei, src_a, dst_a);
  zero_cnt_kernel<<<1, 256, 0, stream>>>(cnt);
  count_kernel<<<cdiv(E2c, 256), 256, 0, stream>>>(dst_a, cnt);
  scan_kernel<<<1, 1, 0, stream>>>(cnt, ofs, fill);
  scatter_kernel<<<cdiv(E2c, 256), 256, 0, stream>>>(dst_a, fill, eix);
  gat_h_kernel<<<cdiv(RR * NN, 256), 256, 0, stream>>>(x, W_gat, att_src, att_dst, hmat, a_s, a_d);
  gat_agg_kernel<<<cdiv(RR * NN * HH, 256), 256, 0, stream>>>(src_a, ofs, eix, a_s, a_d, hmat, b_gat, g0);

  // --- Projector: g1 = relu(g0 @ W_proj + b_proj), bf16 out ---
  launch_gemm(stream, g0, Wproj_pk, RR, 3200, 12800, 0, b_proj, nullptr, 0, nullptr, g1, 0);

  // --- LSTM layer 0: X0 = g1 @ Wih0^T + bih0 + bhh0 (all timesteps) ---
  launch_gemm(stream, g1, Wih0_pk, RR, 2048, 3200, 1, bih0, bhh0, 0, X0, nullptr, 0);
  init_state_kernel<<<cdiv(BB * LHD, 256), 256, 0, stream>>>(c0, c1, h0b, h1b);
  for (int t = 0; t < SS; ++t) {
    launch_gemm(stream, h0b, Whh0_pk, BB, 2048, LHD, 2, X0 + t * 2048, nullptr, SS * 2048,
                gates, nullptr, 0);
    lstm_cell_kernel<<<cdiv(BB * LHD, 256), 256, 0, stream>>>(gates, c0, h0b, H1, t);
  }

  // --- LSTM layer 1 ---
  launch_gemm(stream, H1, Wih1_pk, RR, 2048, LHD, 1, bih1, bhh1, 0, X1, nullptr, 0);
  for (int t = 0; t < SS; ++t) {
    launch_gemm(stream, h1b, Whh1_pk, BB, 2048, LHD, 2, X1 + t * 2048, nullptr, SS * 2048,
                gates, nullptr, 0);
    lstm_cell_kernel<<<cdiv(BB * LHD, 256), 256, 0, stream>>>(gates, c1, h1b, nullptr, t);
  }

  // --- Output head: out = h1(last) @ W_out + b_out ---
  launch_gemm(stream, h1b, Wout_pk, BB, 256, LHD, 3, b_out, nullptr, 0, out, nullptr, OUTD);
}
